// LightweightMambaMoE_66116726555029
// MI455X (gfx1250) — matrix operand order repaired and checked by reading, not compiled, not run on hardware
//
#include <hip/hip_runtime.h>
#include <hip/hip_bf16.h>
#include <math.h>

#define B_   4
#define L_   1024
#define DM   1024
#define DI_  2048
#define DS_  16
#define DC_  4
#define DTR_ 64
#define NL_  2
#define E_   8
#define H_   1024
#define NTOK (B_ * L_)

typedef __attribute__((ext_vector_type(16))) _Float16 v16h;
typedef __attribute__((ext_vector_type(8)))  float    v8f;

__device__ __forceinline__ float sigmoidf_(float x) { return 1.f / (1.f + __expf(-x)); }

// A fragment (16x32 f16): elem i -> K = base + kh*16 + i (contiguous)
__device__ __forceinline__ v16h load_afrag(const float* __restrict__ ap)
{
    v16h f;
    #pragma unroll
    for (int j = 0; j < 8; ++j) {
        f[j]     = (_Float16)ap[j];
        f[8 + j] = (_Float16)ap[8 + j];
    }
    return f;
}
// B fragment (32x16 f16): lane = column, elem i -> K = base + kh*16 + i (contiguous)
__device__ __forceinline__ v16h load_bfrag(const float* __restrict__ wp)
{
    v16h f;
    #pragma unroll
    for (int j = 0; j < 16; ++j) f[j] = (_Float16)wp[j];
    return f;
}

// ---------------------------------------------------------------------------
// Generic WMMA GEMM: C[m,n] = act( sum_k A[row(m),k] * W[n,k] + bias[n] ) * scale(row(m))
//   One wave per 32x32 output tile: 2 A-frags x 2 B-frags -> 4x
//   v_wmma_f32_16x16x32_f16 per K-step (fp32 accumulation).
//   Out-of-range rows/cols are CLAMPED (not branched): clamped lanes compute
//   duplicate rows/cols whose outputs are never stored, keeping the K-loop
//   branch-free and EXEC all-1s for every WMMA.
//   m_count: optional device M (MoE routed counts) -> wave-uniform tile exit.
//   act: 0=none, 1=relu, 2=softplus
// ---------------------------------------------------------------------------
__global__ __launch_bounds__(32)
void gemm_wmma_kernel(const float* __restrict__ A, int lda,
                      const float* __restrict__ W, int ldw,
                      const float* __restrict__ bias,
                      float* __restrict__ C, int ldc,
                      int M, int N, int K,
                      const int* __restrict__ row_idx,
                      const int* __restrict__ m_count,
                      const float* __restrict__ row_scale,
                      int act)
{
    const int tileN = blockIdx.x * 32;
    const int tileM = blockIdx.y * 32;
    const int Mval  = m_count ? *m_count : M;
    if (tileM >= Mval || tileN >= N) return;   // wave-uniform exit

    const int lane = threadIdx.x & 31;
    const int kh   = lane >> 4;     // 0 = lanes 0-15, 1 = lanes 16-31
    const int mn   = lane & 15;

    long aoff[2];
    #pragma unroll
    for (int mt = 0; mt < 2; ++mt) {
        int r = tileM + mt * 16 + mn;
        if (r > Mval - 1) r = Mval - 1;               // clamp, don't branch
        const int t = row_idx ? row_idx[r] : r;
        aoff[mt] = (long)t * lda + kh * 16;
    }
    long woff[2];
    #pragma unroll
    for (int nt = 0; nt < 2; ++nt) {
        int c = tileN + nt * 16 + mn;
        if (c > N - 1) c = N - 1;                     // clamp, don't branch
        woff[nt] = (long)c * ldw + kh * 16;
    }

    v8f acc[2][2] = {};
    for (int kb = 0; kb < K; kb += 32) {
        const v16h a0 = load_afrag(A + aoff[0] + kb);
        const v16h a1 = load_afrag(A + aoff[1] + kb);
        const v16h b0 = load_bfrag(W + woff[0] + kb);
        const v16h b1 = load_bfrag(W + woff[1] + kb);
        if (kb + 32 < K) {                            // uniform branch
            __builtin_prefetch(A + aoff[0] + kb + 32, 0, 0);
            __builtin_prefetch(A + aoff[1] + kb + 32, 0, 0);
            __builtin_prefetch(W + woff[0] + kb + 32, 0, 0);
            __builtin_prefetch(W + woff[1] + kb + 32, 0, 0);
        }
        acc[0][0] = __builtin_amdgcn_wmma_f32_16x16x32_f16(
            false, a0, false, b0, (short)0, acc[0][0], false, false);
        acc[0][1] = __builtin_amdgcn_wmma_f32_16x16x32_f16(
            false, a0, false, b1, (short)0, acc[0][1], false, false);
        acc[1][0] = __builtin_amdgcn_wmma_f32_16x16x32_f16(
            false, a1, false, b0, (short)0, acc[1][0], false, false);
        acc[1][1] = __builtin_amdgcn_wmma_f32_16x16x32_f16(
            false, a1, false, b1, (short)0, acc[1][1], false, false);
    }

    // C/D layout: lane = column n; element r -> row m = subtileM + r + 8*kh
    #pragma unroll
    for (int nt = 0; nt < 2; ++nt) {
        const int n = tileN + nt * 16 + mn;
        if (n < N) {
            const float bv = bias ? bias[n] : 0.f;
            #pragma unroll
            for (int mt = 0; mt < 2; ++mt) {
                #pragma unroll
                for (int r = 0; r < 8; ++r) {
                    const int mrow = tileM + mt * 16 + r + 8 * kh;
                    if (mrow < Mval) {
                        float v = acc[mt][nt][r] + bv;
                        if (act == 1)      v = fmaxf(v, 0.f);
                        else if (act == 2) v = (v > 20.f) ? v : log1pf(expf(v));
                        const int orow = row_idx ? row_idx[mrow] : mrow;
                        if (row_scale) v *= row_scale[orow];
                        C[(long)orow * ldc + n] = v;
                    }
                }
            }
        }
    }
}

// --------------------------- elementwise kernels ---------------------------

__global__ __launch_bounds__(256)
void embed_kernel(const int* __restrict__ tokens, const float* __restrict__ emb,
                  const float* __restrict__ pos, float* __restrict__ x, long n)
{
    long i = (long)blockIdx.x * 256 + threadIdx.x;
    if (i >= n) return;
    long tok = i / DM; int d = (int)(i % DM);
    int tv = tokens[tok];
    int t  = (int)(tok % L_);
    x[i] = emb[(long)tv * DM + d] + pos[(long)t * DM + d];
}

// causal depthwise conv over time + bias + silu; input = u half of uz [N, 2*DI]
__global__ __launch_bounds__(256)
void conv_silu_kernel(const float* __restrict__ uz, const float* __restrict__ cw,
                      const float* __restrict__ cb, float* __restrict__ u, long n)
{
    long i = (long)blockIdx.x * 256 + threadIdx.x;
    if (i >= n) return;
    long tok = i / DI_; int d = (int)(i % DI_);
    int b = (int)(tok / L_), t = (int)(tok % L_);
    float acc = cb[d];
    #pragma unroll
    for (int j = 0; j < DC_; ++j) {
        int tp = t - (DC_ - 1) + j;
        if (tp >= 0)
            acc += cw[d * DC_ + j] * uz[((long)b * L_ + tp) * (2 * DI_) + d];
    }
    u[i] = acc * sigmoidf_(acc);
}

// selective scan: one thread per (b,d) channel, DS states in registers
__global__ __launch_bounds__(256)
void scan_kernel(const float* __restrict__ u, const float* __restrict__ dt,
                 const float* __restrict__ xp, const float* __restrict__ A_log,
                 const float* __restrict__ Dp, float* __restrict__ y)
{
    const int d = blockIdx.x * 256 + threadIdx.x;
    const int b = blockIdx.y;
    float A[DS_], h[DS_];
    #pragma unroll
    for (int s = 0; s < DS_; ++s) { A[s] = -__expf(A_log[d * DS_ + s]); h[s] = 0.f; }
    const float Dv = Dp[d];
    for (int t = 0; t < L_; ++t) {
        const long tok = (long)b * L_ + t;
        const float dtv = dt[tok * DI_ + d];
        const float uv  = u[tok * DI_ + d];
        const float* xr = xp + tok * 96;          // [dt(64) | B(16) | C(16)]
        const float du  = dtv * uv;
        float yv = 0.f;
        #pragma unroll
        for (int s = 0; s < DS_; ++s) {
            h[s] = __expf(dtv * A[s]) * h[s] + du * xr[64 + s];
            yv  += h[s] * xr[80 + s];
        }
        y[tok * DI_ + d] = yv + uv * Dv;
    }
}

// y *= silu(z), z = uz[:, DI:2DI]
__global__ __launch_bounds__(256)
void mulz_kernel(const float* __restrict__ uz, float* __restrict__ y, long n)
{
    long i = (long)blockIdx.x * 256 + threadIdx.x;
    if (i >= n) return;
    long tok = i / DI_; int d = (int)(i % DI_);
    float z = uz[tok * (2 * DI_) + DI_ + d];
    y[i] *= z * sigmoidf_(z);
}

// x[tok] += LayerNorm(src[tok]) * g + b   (Dm = 1024, block = 256, 4 elems/thread)
__global__ __launch_bounds__(256)
void ln_residual_kernel(const float* __restrict__ src, const float* __restrict__ g,
                        const float* __restrict__ bt, float* __restrict__ x)
{
    __shared__ float red[256];
    __shared__ float s_mean, s_rstd;
    const long tok = blockIdx.x;
    const float* row = src + tok * DM;
    float loc[4]; float s = 0.f;
    #pragma unroll
    for (int i = 0; i < 4; ++i) { loc[i] = row[threadIdx.x + i * 256]; s += loc[i]; }
    red[threadIdx.x] = s; __syncthreads();
    for (int off = 128; off > 0; off >>= 1) {
        if (threadIdx.x < off) red[threadIdx.x] += red[threadIdx.x + off];
        __syncthreads();
    }
    if (threadIdx.x == 0) s_mean = red[0] / (float)DM;
    __syncthreads();
    const float mean = s_mean; float v = 0.f;
    #pragma unroll
    for (int i = 0; i < 4; ++i) { float dd = loc[i] - mean; v += dd * dd; }
    red[threadIdx.x] = v; __syncthreads();
    for (int off = 128; off > 0; off >>= 1) {
        if (threadIdx.x < off) red[threadIdx.x] += red[threadIdx.x + off];
        __syncthreads();
    }
    if (threadIdx.x == 0) s_rstd = rsqrtf(red[0] / (float)DM + 1e-5f);
    __syncthreads();
    const float rstd = s_rstd;
    #pragma unroll
    for (int i = 0; i < 4; ++i) {
        const int c = threadIdx.x + i * 256;
        x[tok * DM + c] += (loc[i] - mean) * rstd * g[c] + bt[c];
    }
}

// softmax over E=8 gate logits, keep top-1 weight and index
__global__ __launch_bounds__(256)
void route_kernel(const float* __restrict__ gate, float* __restrict__ top_w,
                  int* __restrict__ top_idx, int N)
{
    int t = blockIdx.x * 256 + threadIdx.x;
    if (t >= N) return;
    const float* gl = gate + (long)t * 8;
    float mx = gl[0];
    #pragma unroll
    for (int j = 1; j < 8; ++j) mx = fmaxf(mx, gl[j]);
    float ev[8], se = 0.f;
    #pragma unroll
    for (int j = 0; j < 8; ++j) { ev[j] = __expf(gl[j] - mx); se += ev[j]; }
    float best = -1.f; int bi = 0;
    #pragma unroll
    for (int j = 0; j < 8; ++j) { float p = ev[j] / se; if (p > best) { best = p; bi = j; } }
    top_w[t] = best; top_idx[t] = bi;
}

__global__ void zero_counts_kernel(int* counts) { if (threadIdx.x < 16) counts[threadIdx.x] = 0; }

__global__ __launch_bounds__(256)
void build_lists_kernel(const int* __restrict__ top_idx, int* __restrict__ counts,
                        int* __restrict__ lists, int N)
{
    int t = blockIdx.x * 256 + threadIdx.x;
    if (t >= N) return;
    int e = top_idx[t];
    int p = atomicAdd(&counts[e], 1);
    lists[(long)e * N + p] = t;
}

__global__ __launch_bounds__(256)
void mask_kernel(const float* __restrict__ x, float* __restrict__ mask)
{
    __shared__ float red[256];
    const long tok = blockIdx.x;
    float s = 0.f;
    #pragma unroll
    for (int i = 0; i < 4; ++i) s += x[tok * DM + threadIdx.x + i * 256];
    red[threadIdx.x] = s; __syncthreads();
    for (int off = 128; off > 0; off >>= 1) {
        if (threadIdx.x < off) red[threadIdx.x] += red[threadIdx.x + off];
        __syncthreads();
    }
    if (threadIdx.x == 0) mask[tok] = (red[0] != 0.f) ? 1.f : 0.f;
}

__global__ __launch_bounds__(256)
void denom_kernel(const float* __restrict__ mask, float* __restrict__ denom)
{
    __shared__ float red[256];
    const int b = blockIdx.x;
    float s = 0.f;
    #pragma unroll
    for (int i = 0; i < 4; ++i) s += mask[b * L_ + threadIdx.x + i * 256];
    red[threadIdx.x] = s; __syncthreads();
    for (int off = 128; off > 0; off >>= 1) {
        if (threadIdx.x < off) red[threadIdx.x] += red[threadIdx.x + off];
        __syncthreads();
    }
    if (threadIdx.x == 0) denom[b] = fmaxf(red[0], 1.f);
}

__global__ __launch_bounds__(256)
void pooled_kernel(const float* __restrict__ x, const float* __restrict__ mask,
                   const float* __restrict__ denom, float* __restrict__ pooled)
{
    const int c = blockIdx.x * 256 + threadIdx.x;
    const int b = blockIdx.y;
    float s = 0.f;
    for (int t = 0; t < L_; ++t)
        s += mask[b * L_ + t] * x[((long)b * L_ + t) * DM + c];
    pooled[b * DM + c] = s / denom[b];
}

// ---------------------------------------------------------------------------

extern "C" void kernel_launch(void* const* d_in, const int* in_sizes, int n_in,
                              void* d_out, int out_size, void* d_ws, size_t ws_size,
                              hipStream_t stream)
{
    (void)in_sizes; (void)n_in; (void)out_size; (void)ws_size;

    const int*   tokens = (const int*)  d_in[0];
    const float* emb    = (const float*)d_in[1];
    const float* pos    = (const float*)d_in[2];
    const float* in_w   = (const float*)d_in[3];
    const float* conv_w = (const float*)d_in[4];
    const float* conv_b = (const float*)d_in[5];
    const float* xp_w   = (const float*)d_in[6];
    const float* dt_w   = (const float*)d_in[7];
    const float* dt_b   = (const float*)d_in[8];
    const float* A_log  = (const float*)d_in[9];
    const float* D_ssm  = (const float*)d_in[10];
    const float* out_w  = (const float*)d_in[11];
    const float* ln1_g  = (const float*)d_in[12];
    const float* ln1_b  = (const float*)d_in[13];
    const float* ln2_g  = (const float*)d_in[14];
    const float* ln2_b  = (const float*)d_in[15];
    const float* gate_w = (const float*)d_in[16];
    const float* gate_b = (const float*)d_in[17];
    const float* e_w1   = (const float*)d_in[18];
    const float* e_b1   = (const float*)d_in[19];
    const float* e_w2   = (const float*)d_in[20];
    const float* e_b2   = (const float*)d_in[21];
    const float* fc1_w  = (const float*)d_in[22];
    const float* fc1_b  = (const float*)d_in[23];
    const float* fc2_w  = (const float*)d_in[24];
    const float* fc2_b  = (const float*)d_in[25];

    // bump allocator over workspace (256B aligned)
    char* cur = (char*)d_ws;
    auto alloc = [&](size_t elems) -> void* {
        void* p = cur; cur += ((elems * 4 + 255) & ~(size_t)255); return p;
    };
    float* xbuf  = (float*)alloc((size_t)NTOK * DM);        // residual stream
    float* uz    = (float*)alloc((size_t)NTOK * 2 * DI_);   // in_proj out [u|z]
    float* ubuf  = (float*)alloc((size_t)NTOK * DI_);       // post-conv silu u
    float* dtf   = (float*)alloc((size_t)NTOK * DI_);       // softplus dt
    float* xpb   = (float*)alloc((size_t)NTOK * 96);        // [dt_r|B|C]
    float* ybuf  = (float*)alloc((size_t)NTOK * DI_);       // scan out / gated
    float* mbuf  = (float*)alloc((size_t)NTOK * DM);        // mamba / moe out
    float* h1    = (float*)alloc((size_t)NTOK * H_);        // expert hidden (token-indexed)
    float* gateb = (float*)alloc((size_t)NTOK * 8);
    float* topw  = (float*)alloc(NTOK);
    float* maskb = (float*)alloc(NTOK);
    float* pooled= (float*)alloc((size_t)B_ * DM);
    float* denom = (float*)alloc(B_);
    float* headh = (float*)alloc((size_t)B_ * 128);
    int*   topi  = (int*)  alloc(NTOK);
    int*   counts= (int*)  alloc(16);
    int*   lists = (int*)  alloc((size_t)E_ * NTOK);

    const dim3 blk32(32), blk256(256);
    const long nXD  = (long)NTOK * DM;
    const long nXDI = (long)NTOK * DI_;

    embed_kernel<<<dim3((unsigned)((nXD + 255) / 256)), blk256, 0, stream>>>(
        tokens, emb, pos, xbuf, nXD);

    for (int l = 0; l < NL_; ++l) {
        const float* in_w_l = in_w  + (size_t)l * 2 * DI_ * DM;
        const float* cw_l   = conv_w + (size_t)l * DI_ * DC_;
        const float* cb_l   = conv_b + (size_t)l * DI_;
        const float* xpw_l  = xp_w  + (size_t)l * (DTR_ + 2 * DS_) * DI_;
        const float* dtw_l  = dt_w  + (size_t)l * DI_ * DTR_;
        const float* dtb_l  = dt_b  + (size_t)l * DI_;
        const float* Al_l   = A_log + (size_t)l * DI_ * DS_;
        const float* Ds_l   = D_ssm + (size_t)l * DI_;
        const float* ow_l   = out_w + (size_t)l * DM * DI_;

        // ---- Mamba ----
        gemm_wmma_kernel<<<dim3((2 * DI_ + 31) / 32, (NTOK + 31) / 32), blk32, 0, stream>>>(
            xbuf, DM, in_w_l, DM, nullptr, uz, 2 * DI_,
            NTOK, 2 * DI_, DM, nullptr, nullptr, nullptr, 0);
        conv_silu_kernel<<<dim3((unsigned)((nXDI + 255) / 256)), blk256, 0, stream>>>(
            uz, cw_l, cb_l, ubuf, nXDI);
        gemm_wmma_kernel<<<dim3((96 + 31) / 32, (NTOK + 31) / 32), blk32, 0, stream>>>(
            ubuf, DI_, xpw_l, DI_, nullptr, xpb, 96,
            NTOK, 96, DI_, nullptr, nullptr, nullptr, 0);
        gemm_wmma_kernel<<<dim3((DI_ + 31) / 32, (NTOK + 31) / 32), blk32, 0, stream>>>(
            xpb, 96, dtw_l, DTR_, dtb_l, dtf, DI_,
            NTOK, DI_, DTR_, nullptr, nullptr, nullptr, 2 /*softplus*/);
        scan_kernel<<<dim3(DI_ / 256, B_), blk256, 0, stream>>>(
            ubuf, dtf, xpb, Al_l, Ds_l, ybuf);
        mulz_kernel<<<dim3((unsigned)((nXDI + 255) / 256)), blk256, 0, stream>>>(
            uz, ybuf, nXDI);
        gemm_wmma_kernel<<<dim3((DM + 31) / 32, (NTOK + 31) / 32), blk32, 0, stream>>>(
            ybuf, DI_, ow_l, DI_, nullptr, mbuf, DM,
            NTOK, DM, DI_, nullptr, nullptr, nullptr, 0);
        ln_residual_kernel<<<dim3(NTOK), blk256, 0, stream>>>(
            mbuf, ln1_g + (size_t)l * DM, ln1_b + (size_t)l * DM, xbuf);

        // ---- MoE (top-1 routed) ----
        gemm_wmma_kernel<<<dim3(1, (NTOK + 31) / 32), blk32, 0, stream>>>(
            xbuf, DM, gate_w + (size_t)l * E_ * DM, DM, gate_b + (size_t)l * E_,
            gateb, 8, NTOK, 8, DM, nullptr, nullptr, nullptr, 0);
        route_kernel<<<dim3(NTOK / 256), blk256, 0, stream>>>(gateb, topw, topi, NTOK);
        zero_counts_kernel<<<dim3(1), blk32, 0, stream>>>(counts);
        build_lists_kernel<<<dim3(NTOK / 256), blk256, 0, stream>>>(topi, counts, lists, NTOK);
        for (int e = 0; e < E_; ++e) {
            const float* w1 = e_w1 + ((size_t)l * E_ + e) * H_ * DM;
            const float* b1 = e_b1 + ((size_t)l * E_ + e) * H_;
            const float* w2 = e_w2 + ((size_t)l * E_ + e) * DM * H_;
            const float* b2 = e_b2 + ((size_t)l * E_ + e) * DM;
            gemm_wmma_kernel<<<dim3((H_ + 31) / 32, (NTOK + 31) / 32), blk32, 0, stream>>>(
                xbuf, DM, w1, DM, b1, h1, H_,
                NTOK, H_, DM, lists + (size_t)e * NTOK, counts + e, nullptr, 1 /*relu*/);
            gemm_wmma_kernel<<<dim3((DM + 31) / 32, (NTOK + 31) / 32), blk32, 0, stream>>>(
                h1, H_, w2, H_, b2, mbuf, DM,
                NTOK, DM, H_, lists + (size_t)e * NTOK, counts + e, topw, 0);
        }
        ln_residual_kernel<<<dim3(NTOK), blk256, 0, stream>>>(
            mbuf, ln2_g + (size_t)l * DM, ln2_b + (size_t)l * DM, xbuf);
    }

    // ---- head: masked mean pool -> fc1 relu -> fc2 ----
    mask_kernel<<<dim3(NTOK), blk256, 0, stream>>>(xbuf, maskb);
    denom_kernel<<<dim3(B_), blk256, 0, stream>>>(maskb, denom);
    pooled_kernel<<<dim3(DM / 256, B_), blk256, 0, stream>>>(xbuf, maskb, denom, pooled);
    gemm_wmma_kernel<<<dim3((128 + 31) / 32, 1), blk32, 0, stream>>>(
        pooled, DM, fc1_w, DM, fc1_b, headh, 128,
        B_, 128, DM, nullptr, nullptr, nullptr, 1 /*relu*/);
    gemm_wmma_kernel<<<dim3(1, 1), blk32, 0, stream>>>(
        headh, 128, fc2_w, 128, fc2_b, (float*)d_out, 2,
        B_, 2, 128, nullptr, nullptr, nullptr, 0);
}
